// FMoE_644245095184
// MI455X (gfx1250) — matrix operand order repaired and checked by reading, not compiled, not run on hardware
//
#include <hip/hip_runtime.h>
#include <hip/hip_bf16.h>
#include <math.h>

typedef __attribute__((ext_vector_type(16))) __bf16 v16bf;
typedef __attribute__((ext_vector_type(8)))  __bf16 v8bf;
typedef __attribute__((ext_vector_type(4)))  __bf16 v4bf;
typedef __attribute__((ext_vector_type(2)))  __bf16 v2bf;
typedef __attribute__((ext_vector_type(8)))  float  v8f;
typedef __attribute__((ext_vector_type(4)))  float  v4f;
typedef __attribute__((ext_vector_type(2)))  float  v2f;

#define T_TOK   2048
#define D_MODEL 1024
#define H_HID   2048
#define NEXP    8
#define HC      512                 // H chunk staged in LDS per iteration
#define DSPLIT  2                   // split D across gridDim.y for 2x parallelism
#define DSLICE  (D_MODEL / DSPLIT)  // 512 output cols per workgroup

// Branch-free tanh-approx gelu using hw v_exp_f32 (2^x) and v_rcp_f32.
__device__ __forceinline__ float gelu_tanh(float x) {
    const float k0 = 0.7978845608028654f;   // sqrt(2/pi)
    const float k1 = 0.044715f;
    float z  = k0 * (x + k1 * x * x * x);
    float ez = __builtin_amdgcn_exp2f(2.8853900817779268f * z); // 2^(2*log2e*z)
    float th = 1.0f - 2.0f * __builtin_amdgcn_rcpf(1.0f + ez);
    return 0.5f * x * (1.0f + th);
}

// Vector fptrunc pair -> single v_cvt_pk_bf16_f32.
__device__ __forceinline__ unsigned pack_bf16x2(float lo, float hi) {
    v2f t; t[0] = lo; t[1] = hi;
    v2bf h = __builtin_convertvector(t, v2bf);
    union { v2bf v; unsigned u; } pk; pk.v = h;
    return pk.u;
}

// Stage a 32(k) x 16(n) fp32 weight block, transposed+converted, into a
// per-wave LDS buffer laid out [n][k] bf16 (k contiguous). Each lane owns
// one k-pair (p) and one n-half (nh): 4x global b128 in, 8x ds_store_b32 out.
__device__ __forceinline__ void stage_weights(
    const float* __restrict__ w00,   // &W[k0][n0], row stride = `stride` floats
    size_t stride, __bf16 (*stg)[32], int lane)
{
    const int p  = lane & 15;        // k-pair index: rows 2p, 2p+1
    const int nh = lane >> 4;        // n half: cols nh*8 .. nh*8+7
    const float* r0 = w00 + (size_t)(2 * p) * stride + nh * 8;
    const float* r1 = r0 + stride;
    float4 a0 = ((const float4*)r0)[0], a1 = ((const float4*)r0)[1];
    float4 b0 = ((const float4*)r1)[0], b1 = ((const float4*)r1)[1];
    float av[8] = {a0.x,a0.y,a0.z,a0.w, a1.x,a1.y,a1.z,a1.w};
    float bv[8] = {b0.x,b0.y,b0.z,b0.w, b1.x,b1.y,b1.z,b1.w};
#pragma unroll
    for (int i = 0; i < 8; ++i)
        *(unsigned*)&stg[nh * 8 + i][2 * p] = pack_bf16x2(av[i], bv[i]);
}

// ---------------------------------------------------------------------------
// Gate: one wave32 per token. logits = x @ Wg + bg ; top-2 ; softmax.
// ---------------------------------------------------------------------------
__global__ void __launch_bounds__(256) moe_gate(
    const float* __restrict__ X, const float* __restrict__ Wg,
    const float* __restrict__ bg, float* __restrict__ wfull,
    int* __restrict__ top_idx)
{
    const int wave = threadIdx.x >> 5;
    const int lane = threadIdx.x & 31;
    const int t = blockIdx.x * 8 + wave;
    if (t >= T_TOK) return;

    float acc[NEXP];
#pragma unroll
    for (int e = 0; e < NEXP; ++e) acc[e] = 0.0f;

    const float* xr = X + (size_t)t * D_MODEL;
    for (int d = lane; d < D_MODEL; d += 32) {
        float xv = xr[d];
        const float4* wg4 = (const float4*)(Wg + (size_t)d * NEXP);
        float4 wa = wg4[0], wb = wg4[1];
        acc[0] += xv * wa.x; acc[1] += xv * wa.y;
        acc[2] += xv * wa.z; acc[3] += xv * wa.w;
        acc[4] += xv * wb.x; acc[5] += xv * wb.y;
        acc[6] += xv * wb.z; acc[7] += xv * wb.w;
    }
#pragma unroll
    for (int e = 0; e < NEXP; ++e)
        for (int off = 16; off > 0; off >>= 1)
            acc[e] += __shfl_down(acc[e], off);

    if (lane == 0) {
        float lg[NEXP];
#pragma unroll
        for (int e = 0; e < NEXP; ++e) lg[e] = acc[e] + bg[e];
        int i0 = 0;
#pragma unroll
        for (int e = 1; e < NEXP; ++e) if (lg[e] > lg[i0]) i0 = e;
        int i1 = (i0 == 0) ? 1 : 0;
#pragma unroll
        for (int e = 0; e < NEXP; ++e)
            if (e != i0 && lg[e] > lg[i1]) i1 = e;
        float ex  = __builtin_amdgcn_exp2f(1.4426950408889634f * (lg[i1] - lg[i0]));
        float inv = __builtin_amdgcn_rcpf(1.0f + ex);
        float s0 = inv, s1 = ex * inv;
#pragma unroll
        for (int e = 0; e < NEXP; ++e)
            wfull[(size_t)t * NEXP + e] = (e == i0) ? s0 : ((e == i1) ? s1 : 0.0f);
        top_idx[2 * t + 0] = i0;
        top_idx[2 * t + 1] = i1;
    }
}

// ---------------------------------------------------------------------------
// Main fused MoE FFN. One workgroup = 16-token tile x 512 output cols.
// bf16 WMMA 16x16x32, f32 accumulate; experts with no routed token skipped.
// ---------------------------------------------------------------------------
__global__ void __launch_bounds__(256) moe_ffn(
    const float* __restrict__ X,  const float* __restrict__ W1,
    const float* __restrict__ b1, const float* __restrict__ W2,
    const float* __restrict__ b2, const float* __restrict__ wfull,
    float* __restrict__ out)
{
    __shared__ __bf16 Xbf[16][D_MODEL];   // 32 KB  token tile (bf16)
    __shared__ __bf16 Hbf[16][HC];        // 16 KB  hidden chunk (w * gelu(...))
    __shared__ __bf16 stage[8][16][32];   //  8 KB  per-wave B-transpose staging
    __shared__ __align__(16) float wlds[16];

    const int tid  = threadIdx.x;
    const int w    = tid >> 5;            // wave id 0..7
    const int lane = tid & 31;
    const int r    = lane & 15;
    const int kh   = lane >> 4;           // K-half selector

    const int m0    = blockIdx.x * 16;
    const int dbase = blockIdx.y * DSLICE;

    // Stage X tile fp32 -> bf16 (b128 in, packed cvt, b64 out).
    for (int idx = tid; idx < 16 * D_MODEL / 4; idx += 256) {
        int row  = idx >> 8;              // 256 float4 per row
        int col4 = idx & 255;
        float4 v = ((const float4*)(X + (size_t)(m0 + row) * D_MODEL))[col4];
        v4f t; t[0] = v.x; t[1] = v.y; t[2] = v.z; t[3] = v.w;
        v4bf h = __builtin_convertvector(t, v4bf);  // 2x v_cvt_pk_bf16_f32
        *(v4bf*)&Xbf[row][col4 * 4] = h;
    }
    __syncthreads();

    // Persistent output accumulators: wave w owns cols [dbase+w*64, +64) = 4 tiles.
    v8f y[4];
#pragma unroll
    for (int t = 0; t < 4; ++t)
#pragma unroll
        for (int j = 0; j < 8; ++j) y[t][j] = 0.0f;

    for (int e = 0; e < NEXP; ++e) {
        if (tid < 16) wlds[tid] = wfull[(size_t)(m0 + tid) * NEXP + e];
        __syncthreads();

        // uniform expert-skip check (4x b128 LDS loads)
        const float4* w4 = (const float4*)wlds;
        float4 c0 = w4[0], c1 = w4[1], c2 = w4[2], c3 = w4[3];
        float s = fabsf(c0.x) + fabsf(c0.y) + fabsf(c0.z) + fabsf(c0.w)
                + fabsf(c1.x) + fabsf(c1.y) + fabsf(c1.z) + fabsf(c1.w)
                + fabsf(c2.x) + fabsf(c2.y) + fabsf(c2.z) + fabsf(c2.w)
                + fabsf(c3.x) + fabsf(c3.y) + fabsf(c3.z) + fabsf(c3.w);
        if (s == 0.0f) { __syncthreads(); continue; }  // uniform decision

        float wreg[8];
#pragma unroll
        for (int j = 0; j < 8; ++j) wreg[j] = wlds[kh * 8 + j];   // token m = kh*8+j

        // out += w * b2[e]  (bias of second GEMM, gate-weighted)
#pragma unroll
        for (int t = 0; t < 4; ++t) {
            float bv = b2[(size_t)e * D_MODEL + dbase + w * 64 + t * 16 + r];
#pragma unroll
            for (int j = 0; j < 8; ++j) y[t][j] += wreg[j] * bv;
        }

        for (int c = 0; c < H_HID / HC; ++c) {   // 4 chunks of 512 hidden cols
            // ---- phase A: Hbf = w * gelu(X @ W1[e][:, chunk] + b1) -------------
#pragma unroll
            for (int t = 0; t < 4; ++t) {
                const int ncol = w * 64 + t * 16;   // col within chunk
                const int nglb = c * HC + ncol;     // col within H
                float b1v = b1[(size_t)e * H_HID + nglb + r];  // hoisted: latency
                v8f acc;                                       // hidden by kc loop
#pragma unroll
                for (int j = 0; j < 8; ++j) acc[j] = 0.0f;

#pragma unroll 2
                for (int kc = 0; kc < D_MODEL / 32; ++kc) {
                    const int k0 = kc * 32;
                    stage_weights(W1 + ((size_t)e * D_MODEL + k0) * H_HID + nglb,
                                  H_HID, stage[w], lane);

                    // A (16x32 bf16): lanes 0-15 K{0..7,16..23}, 16-31 K{8..15,24..31}
                    v8bf alo = *(const v8bf*)&Xbf[r][k0 + kh * 8];
                    v8bf ahi = *(const v8bf*)&Xbf[r][k0 + 16 + kh * 8];
                    v16bf A = __builtin_shufflevector(alo, ahi,
                        0,1,2,3,4,5,6,7,8,9,10,11,12,13,14,15);
                    // B (32x16 bf16): lanes 0-15 K{0..7,16..23}, 16-31 K{8..15,24..31}
                    v8bf blo = *(const v8bf*)&stage[w][r][kh * 8];
                    v8bf bhi = *(const v8bf*)&stage[w][r][16 + kh * 8];
                    v16bf B = __builtin_shufflevector(blo, bhi,
                        0,1,2,3,4,5,6,7,8,9,10,11,12,13,14,15);

                    acc = __builtin_amdgcn_wmma_f32_16x16x32_bf16(
                        false, A, false, B, (short)0, acc, false, false);
                }
                // epilogue: +b1, gelu, gate weight, bf16 store to Hbf
#pragma unroll
                for (int j = 0; j < 8; ++j) {
                    int m = kh * 8 + j;
                    float hv = gelu_tanh(acc[j] + b1v) * wreg[j];
                    Hbf[m][ncol + r] = (__bf16)hv;
                }
            }
            __syncthreads();   // Hbf chunk complete

            // ---- phase B: y += Hbf_chunk @ W2[e][chunk, dslice] ----------------
#pragma unroll
            for (int t = 0; t < 4; ++t) {
                const int n0 = dbase + w * 64 + t * 16;
#pragma unroll 2
                for (int kc = 0; kc < HC / 32; ++kc) {
                    const int h0 = c * HC + kc * 32;
                    stage_weights(W2 + ((size_t)e * H_HID + h0) * D_MODEL + n0,
                                  D_MODEL, stage[w], lane);

                    v8bf alo = *(const v8bf*)&Hbf[r][kc * 32 + kh * 8];
                    v8bf ahi = *(const v8bf*)&Hbf[r][kc * 32 + 16 + kh * 8];
                    v16bf A = __builtin_shufflevector(alo, ahi,
                        0,1,2,3,4,5,6,7,8,9,10,11,12,13,14,15);
                    v8bf blo = *(const v8bf*)&stage[w][r][kh * 8];
                    v8bf bhi = *(const v8bf*)&stage[w][r][16 + kh * 8];
                    v16bf B = __builtin_shufflevector(blo, bhi,
                        0,1,2,3,4,5,6,7,8,9,10,11,12,13,14,15);

                    y[t] = __builtin_amdgcn_wmma_f32_16x16x32_bf16(
                        false, A, false, B, (short)0, y[t], false, false);
                }
            }
            __syncthreads();   // before next chunk overwrites Hbf
        }
    }

    // Store output tile (each workgroup owns its region exclusively).
#pragma unroll
    for (int t = 0; t < 4; ++t) {
        int d = dbase + w * 64 + t * 16 + r;
#pragma unroll
        for (int j = 0; j < 8; ++j) {
            int m = kh * 8 + j;
            out[(size_t)(m0 + m) * D_MODEL + d] = y[t][j];
        }
    }
}

// ---------------------------------------------------------------------------
extern "C" void kernel_launch(void* const* d_in, const int* in_sizes, int n_in,
                              void* d_out, int out_size, void* d_ws, size_t ws_size,
                              hipStream_t stream)
{
    const float* X  = (const float*)d_in[0];
    const float* Wg = (const float*)d_in[1];
    const float* bg = (const float*)d_in[2];
    const float* W1 = (const float*)d_in[3];
    const float* b1 = (const float*)d_in[4];
    const float* W2 = (const float*)d_in[5];
    const float* b2 = (const float*)d_in[6];
    // d_in[7] = top_k (always 2 here)

    float* out     = (float*)d_out;
    int*   top_idx = (int*)d_out + (size_t)T_TOK * D_MODEL;  // tuple tail
    float* wfull   = (float*)d_ws;                           // T*E gate weights

    moe_gate<<<T_TOK / 8, 256, 0, stream>>>(X, Wg, bg, wfull, top_idx);

    dim3 grid(T_TOK / 16, DSPLIT);
    moe_ffn<<<grid, 256, 0, stream>>>(X, W1, b1, W2, b2, wfull, out);
}
